// HebbianRNN_40114994544685
// MI455X (gfx1250) — matrix operand order repaired and checked by reading, not compiled, not run on hardware
//
#include <hip/hip_runtime.h>

// ---------------------------------------------------------------------------
// Hebbian plastic RNN, fully fused, LDS-resident state (gfx1250 / MI455X).
// One workgroup (512 thr = 16 waves) per batch element. hebb + Wh live in LDS
// as bf16 for the entire T=1024 loop -> ~260MB total HBM traffic instead of
// ~256GB. Matvecs, the in-chunk correction combine, and the chunked rank-32
// Hebbian update all run on v_wmma_f32_16x16x32_bf16.
// ---------------------------------------------------------------------------

typedef __attribute__((ext_vector_type(16))) __bf16       v16bf;
typedef __attribute__((ext_vector_type(8)))  float        v8f;
typedef __attribute__((ext_vector_type(4)))  unsigned int u32x4;

union ABOp { u32x4 q[2]; v16bf v; };

constexpr int   T_   = 1024;
constexpr int   DIN  = 64;
constexpr int   H    = 256;
constexpr int   DOUT = 64;
constexpr float ETA  = 0.1f;
constexpr float DEC  = 0.9f;
constexpr float D32  = 0.03433683820292512f;   // 0.9^32
constexpr float D31  = 0.03815204244769458f;   // 0.9^31
constexpr float INV_D   = 1.0f / 0.9f;
constexpr float INV_D32 = 1.0f / D32;

// LDS carve (bytes). 16B aligned everywhere. Total 305296 < 320KB.
constexpr unsigned OFF_WHS  = 0;                       // Wh  bf16 swizzled  128KB
constexpr unsigned OFF_HBS  = OFF_WHS  + 131072;       // hebb bf16 swizzled 128KB
constexpr unsigned OFF_ACH  = OFF_HBS  + 131072;       // a-history [256][32] bf16
constexpr unsigned OFF_BHH  = OFF_ACH  + 16384;        // b-history [256][32] bf16
constexpr unsigned OFF_HF   = OFF_BHH  + 16384;        // h fp32 [256]
constexpr unsigned OFF_HBF  = OFF_HF   + 1024;         // h bf16 [256]
constexpr unsigned OFF_SX   = OFF_HBF  + 512;          // x_t [64]
constexpr unsigned OFF_XP   = OFF_SX   + 256;          // xin partials [2][256]
constexpr unsigned OFF_SWH  = OFF_XP   + 2048;         // Wh@h  [256]
constexpr unsigned OFF_SHB  = OFF_SWH  + 1024;         // hebb0@h [256]
constexpr unsigned OFF_SCOR = OFF_SHB  + 1024;         // correction vec [256] f32
constexpr unsigned OFF_CDBF = OFF_SCOR + 1024;         // correction dots [32] bf16
constexpr unsigned OFF_OP   = OFF_CDBF + 128;          // out partials [8][64]
constexpr unsigned OFF_BSH  = OFF_OP   + 2048;         // bi+bh [256]
constexpr unsigned OFF_BSO  = OFF_BSH  + 1024;         // bo [64]
constexpr unsigned OFF_SMD  = OFF_BSO  + 256;          // mod scalar
constexpr unsigned SMEM_BYTES = OFF_SMD + 16;          // 305296

__device__ __forceinline__ float bf2f(unsigned short h) {
  return __builtin_bit_cast(float, (unsigned int)h << 16);
}
__device__ __forceinline__ unsigned short f2bf(float f) {
  unsigned int u = __builtin_bit_cast(unsigned int, f);
  u = u + 0x7FFFu + ((u >> 16) & 1u);            // RNE
  return (unsigned short)(u >> 16);
}
// physical dword index for logical (row, dword-col) with per-row rotation so
// 16-row WMMA strip loads are bank-conflict free (64 banks, 128 dw per row).
__device__ __forceinline__ int swz(int row, int jdw) {
  return row * 128 + ((jdw + 4 * row) & 127);
}
// Branchless B-operand "column 0" build: all 16 lanes of each half broadcast-
// load the same 32 bytes (conflict-free), then AND with a lane mask so only
// lanes 0 and 16 keep data (bf16 zero == 0x0000). Avoids EXEC save/restore
// and mov-zero bursts in the hot loop.
__device__ __forceinline__ void load_bcol(const unsigned short* p, unsigned msk,
                                          ABOp& bb) {
  u32x4 m = {msk, msk, msk, msk};
  bb.q[0] = (*(const u32x4*)(p))     & m;
  bb.q[1] = (*(const u32x4*)(p + 16)) & m;
}

__global__ __launch_bounds__(512)
void hebb_rnn_fused(const float* __restrict__ x,  const float* __restrict__ mod,
                    const float* __restrict__ Wi, const float* __restrict__ bi,
                    const float* __restrict__ Wh, const float* __restrict__ bh,
                    const float* __restrict__ Wo, const float* __restrict__ bo,
                    float* __restrict__ out)
{
  extern __shared__ char smem[];
  unsigned int*   whs  = (unsigned int*)  (smem + OFF_WHS);
  unsigned int*   hbs  = (unsigned int*)  (smem + OFF_HBS);
  unsigned short* wh16 = (unsigned short*)(smem + OFF_WHS);
  unsigned short* hb16 = (unsigned short*)(smem + OFF_HBS);
  unsigned short* ach  = (unsigned short*)(smem + OFF_ACH);
  unsigned short* bhh  = (unsigned short*)(smem + OFF_BHH);
  float*          h_f  = (float*)         (smem + OFF_HF);
  unsigned short* h_bf = (unsigned short*)(smem + OFF_HBF);
  float*          sx   = (float*)         (smem + OFF_SX);
  float*          xp   = (float*)         (smem + OFF_XP);
  float*          swh  = (float*)         (smem + OFF_SWH);
  float*          shb  = (float*)         (smem + OFF_SHB);
  float*          scor = (float*)         (smem + OFF_SCOR);
  unsigned short* cDbf = (unsigned short*)(smem + OFF_CDBF);
  float*          op   = (float*)         (smem + OFF_OP);
  float*          bsh  = (float*)         (smem + OFF_BSH);
  float*          bso  = (float*)         (smem + OFF_BSO);
  float*          smd  = (float*)         (smem + OFF_SMD);

  const int tid  = threadIdx.x;
  const int lane = tid & 31;
  const int w    = tid >> 5;                 // wave 0..15
  const int b    = blockIdx.x;
  const int ln15 = lane & 15;
  const int lh   = lane >> 4;
  const unsigned msk = (ln15 == 0) ? 0xFFFFFFFFu : 0u;

  // ---------------- preamble: stage weights / zero state ----------------
  for (int idx = tid; idx < H * H; idx += 512) {      // Wh -> bf16, swizzled
    int r = idx >> 8, c = idx & 255;
    wh16[swz(r, c >> 1) * 2 + (c & 1)] = f2bf(Wh[idx]);
  }
  for (int idx = tid; idx < 32768; idx += 512) hbs[idx] = 0u;       // hebb = 0
  for (int idx = tid; idx < 4096;  idx += 512) {                    // history = 0
    ((unsigned int*)ach)[idx] = 0u; ((unsigned int*)bhh)[idx] = 0u;
  }
  if (tid < H)    { bsh[tid] = bi[tid] + bh[tid]; h_f[tid] = 0.0f; h_bf[tid] = 0; }
  if (tid < DOUT) { bso[tid] = bo[tid]; }

  // register-resident Wi / Wo slices (loaded once, reused 1024x)
  float wir[32];
  { int i = tid & 255, hf2 = tid >> 8;
    #pragma unroll
    for (int c = 0; c < 32; ++c) wir[c] = Wi[i * DIN + hf2 * 32 + c]; }
  float wor[32];
  { int o = tid & 63, pt = tid >> 6;
    #pragma unroll
    for (int c = 0; c < 32; ++c) wor[c] = Wo[o * H + pt * 32 + c]; }

  __syncthreads();

  const float* xg = x   + (size_t)b * T_ * DIN;
  const float* mg = mod + (size_t)b * T_;
  float*       og = out + (size_t)b * T_ * DOUT;

  float dk  = 1.0f;          // 0.9^kk
  float asc = ETA * D31;     // eta * 0.9^(31-kk)

  for (int t = 0; t < T_; ++t) {
    const int kk = t & 31;
    if (kk == 0) { dk = 1.0f; asc = ETA * D31; }

    // ---- P1: stage x_t, mod_t; prefetch next x_t ----
    if (tid < DIN) sx[tid] = xg[t * DIN + tid];
    if (tid == 0) {
      smd[0] = mg[t];
      if (t + 1 < T_) __builtin_prefetch(xg + (t + 1) * DIN, 0, 0);
    }
    __syncthreads();

    // ---- P2: xin partials, a-history write, WMMA matvecs, correction dots ----
    { int i = tid & 255, hf2 = tid >> 8;
      float p = 0.0f;
      #pragma unroll
      for (int c = 0; c < 32; ++c) p += wir[c] * sx[hf2 * 32 + c];
      xp[hf2 * 256 + i] = p; }
    if (tid < H) ach[tid * 32 + kk] = f2bf(asc * smd[0] * h_f[tid]);

    { // dual matvec: rows 16w..16w+15, (Wh@h) and (hebb0@h), h in B column 0
      v8f accW = {0,0,0,0,0,0,0,0};
      v8f accH = {0,0,0,0,0,0,0,0};
      const int row = 16 * w + ln15;
      #pragma unroll
      for (int c = 0; c < 8; ++c) {
        const int jb   = 32 * c;
        const int jdwb = (jb >> 1) + lh * 4;
        const int p1 = swz(row, jdwb);
        const int p2 = swz(row, jdwb + 8);
        ABOp a1; a1.q[0] = *(const u32x4*)(whs + p1); a1.q[1] = *(const u32x4*)(whs + p2);
        ABOp a2; a2.q[0] = *(const u32x4*)(hbs + p1); a2.q[1] = *(const u32x4*)(hbs + p2);
        ABOp bb; load_bcol(h_bf + jb + lh * 8, msk, bb);
        accW = __builtin_amdgcn_wmma_f32_16x16x32_bf16(false, a1.v, false, bb.v,
                                                       (short)0, accW, false, false);
        accH = __builtin_amdgcn_wmma_f32_16x16x32_bf16(false, a2.v, false, bb.v,
                                                       (short)0, accH, false, false);
      }
      if (ln15 == 0) {
        const int mb = 16 * w + lh * 8;
        #pragma unroll
        for (int v = 0; v < 8; ++v) { swh[mb + v] = accW[v]; shb[mb + v] = accH[v]; }
      }
    }
    { // correction dots D_s = b_s . h  (unwritten slots are zero), packed bf16
      float pa = 0.0f, pb = 0.0f;
      #pragma unroll
      for (int jj = 0; jj < 8; ++jj) {
        const int i = lane + 32 * jj;
        const float hv = h_f[i];
        pa += bf2f(bhh[i * 32 + w])      * hv;
        pb += bf2f(bhh[i * 32 + w + 16]) * hv;
      }
      #pragma unroll
      for (int off = 16; off > 0; off >>= 1) {
        pa += __shfl_down(pa, off, 32);
        pb += __shfl_down(pb, off, 32);
      }
      if (lane == 0) { cDbf[w] = f2bf(pa); cDbf[w + 16] = f2bf(pb); }
    }
    __syncthreads();

    // ---- P2b: correction combine corr = Achist (256x32) . D (32) via WMMA ----
    {
      ABOp Ac;
      { const int row = 16 * w + ln15; const int sb = lh * 8;
        Ac.q[0] = *(const u32x4*)(ach + row * 32 + sb);
        Ac.q[1] = *(const u32x4*)(ach + row * 32 + 16 + sb); }
      ABOp Bc; load_bcol(cDbf + lh * 8, msk, Bc);
      v8f z8 = {0,0,0,0,0,0,0,0};
      v8f cr = __builtin_amdgcn_wmma_f32_16x16x32_bf16(false, Ac.v, false, Bc.v,
                                                       (short)0, z8, false, false);
      if (ln15 == 0) {
        const int mb = 16 * w + lh * 8;
        #pragma unroll
        for (int v = 0; v < 8; ++v) scor[mb + v] = cr[v];
      }
    }
    __syncthreads();

    // ---- P3: combine + tanh -> new h ----
    if (tid < H) {
      const int i = tid;
      const float pre = xp[i] + xp[256 + i] + bsh[i] + swh[i]
                      + dk * shb[i] + (dk * INV_D32) * scor[i];
      const float hn = tanhf(pre);
      h_f[i] = hn;
      const unsigned short hb = f2bf(hn);
      h_bf[i] = hb;
      bhh[i * 32 + kk] = hb;
    }
    __syncthreads();

    // ---- P4: output projection partials ----
    { const int o = tid & 63, pt = tid >> 6;
      float p = 0.0f;
      #pragma unroll
      for (int c = 0; c < 32; ++c) p += wor[c] * h_f[pt * 32 + c];
      op[pt * 64 + o] = p; }
    __syncthreads();

    // ---- P5: reduce + store out[b,t,:] ----
    if (tid < DOUT) {
      float s = bso[tid];
      #pragma unroll
      for (int pt = 0; pt < 8; ++pt) s += op[pt * 64 + tid];
      og[t * DOUT + tid] = s;
    }

    // ---- P6: end-of-chunk rank-32 Hebbian writeback via WMMA ----
    if (kk == 31) {                       // uniform across block
      __syncthreads();
      ABOp Aop;                           // A[m][s] = a_s[16w+m]
      { const int row = 16 * w + ln15; const int sb = lh * 8;
        Aop.q[0] = *(const u32x4*)(ach + row * 32 + sb);
        Aop.q[1] = *(const u32x4*)(ach + row * 32 + 16 + sb); }
      for (int tj = 0; tj < 16; ++tj) {
        ABOp Bop;                         // B[s][n] = b_s[16tj+n]
        { const int col = 16 * tj + ln15; const int kb = lh * 8;
          Bop.q[0] = *(const u32x4*)(bhh + col * 32 + kb);
          Bop.q[1] = *(const u32x4*)(bhh + col * 32 + kb + 16); }
        const int colc = 16 * tj + ln15;
        const int jdw = colc >> 1, hlo = colc & 1;
        v8f cc;                            // C = 0.9^32 * hebb0 tile
        #pragma unroll
        for (int v = 0; v < 8; ++v) {
          const int r = 16 * w + v + 8 * lh;
          cc[v] = D32 * bf2f(hb16[swz(r, jdw) * 2 + hlo]);
        }
        v8f dd = __builtin_amdgcn_wmma_f32_16x16x32_bf16(false, Aop.v, false, Bop.v,
                                                         (short)0, cc, false, false);
        #pragma unroll
        for (int v = 0; v < 8; ++v) {
          const int r = 16 * w + v + 8 * lh;
          hb16[swz(r, jdw) * 2 + hlo] = f2bf(dd[v]);
        }
      }
      __syncthreads();                    // all history reads done
      for (int idx = tid; idx < 4096; idx += 512) {
        ((unsigned int*)ach)[idx] = 0u; ((unsigned int*)bhh)[idx] = 0u;
      }
    }
    __syncthreads();

    dk  *= DEC;
    asc *= INV_D;
  }
}

extern "C" void kernel_launch(void* const* d_in, const int* in_sizes, int n_in,
                              void* d_out, int out_size, void* d_ws, size_t ws_size,
                              hipStream_t stream) {
  (void)n_in; (void)out_size; (void)d_ws; (void)ws_size;
  const float* x  = (const float*)d_in[0];
  const float* md = (const float*)d_in[1];
  const float* Wi = (const float*)d_in[2];
  const float* bi = (const float*)d_in[3];
  const float* Wh = (const float*)d_in[4];
  const float* bh = (const float*)d_in[5];
  const float* Wo = (const float*)d_in[6];
  const float* bo = (const float*)d_in[7];
  float* o = (float*)d_out;

  const int B = in_sizes[1] / T_;   // mod_signal_seq is [B,T]

  (void)hipFuncSetAttribute(reinterpret_cast<const void*>(hebb_rnn_fused),
                            hipFuncAttributeMaxDynamicSharedMemorySize,
                            (int)SMEM_BYTES);
  hebb_rnn_fused<<<B, 512, SMEM_BYTES, stream>>>(x, md, Wi, bi, Wh, bh, Wo, bo, o);
}
